// RelNNTransformConnected_77841987273099
// MI455X (gfx1250) — matrix operand order repaired and checked by reading, not compiled, not run on hardware
//
#include <hip/hip_runtime.h>
#include <hip/hip_bf16.h>

typedef _Float16 f16;
typedef __attribute__((ext_vector_type(8)))  f16   v8h;
typedef __attribute__((ext_vector_type(16))) f16   v16h;
typedef __attribute__((ext_vector_type(8)))  float v8f;

#define B_   64
#define N_   128
#define FIN_ 8
#define H_   128

// tanh(x) = 1 - 2/(exp(2x)+1); exp2-based, saturates correctly via inf/0.
__device__ __forceinline__ float fast_tanh(float x) {
  float e2 = __builtin_amdgcn_exp2f(x * 2.8853900817779268f); // 2*log2(e)
  return 1.0f - 2.0f * __builtin_amdgcn_rcpf(e2 + 1.0f);
}

__device__ __forceinline__ v16h cat8(v8h lo, v8h hi) {
  return __builtin_shufflevector(lo, hi, 0,1,2,3,4,5,6,7,8,9,10,11,12,13,14,15);
}

// One block per batch. 8 waves, each owns one 16-row M-tile of the 128x128 output.
__global__ __launch_bounds__(256)
void node_mlp_kernel(const float* __restrict__ jets,  // [B,N,FIN]
                     const float* __restrict__ Wu,    // [H,FIN]
                     const float* __restrict__ bu,    // [H]
                     const float* __restrict__ Wu1,   // [H,H]
                     const float* __restrict__ bu1,   // [H]
                     const float* __restrict__ We,    // [H,H]
                     float* __restrict__ eOut)        // [B,N,H]
{
  __shared__ f16 sX0[N_ * H_];   // 32 KB
  __shared__ f16 sX1[N_ * H_];   // 32 KB
  __shared__ f16 sW [H_ * H_];   // 32 KB (holds W_u1, then W_e)

  const int b    = blockIdx.x;
  const int tid  = threadIdx.x;
  const int wave = tid >> 5;
  const int lane = tid & 31;
  const int half = lane >> 4;     // 0: lanes 0-15, 1: lanes 16-31
  const int l16  = lane & 15;

  // ---- Stage 0: stash W_u1 (f16) and compute x0 = relu(jets*Wu^T + bu) ----
  for (int i = tid; i < H_ * H_; i += 256) sW[i] = (f16)Wu1[i];
  const float* jb = jets + (size_t)b * N_ * FIN_;
  for (int i = tid; i < N_ * H_; i += 256) {
    int n = i >> 7, h = i & 127;
    float acc = bu[h];
    #pragma unroll
    for (int f = 0; f < FIN_; ++f) acc += jb[n * FIN_ + f] * Wu[h * FIN_ + f];
    sX0[i] = (f16)fmaxf(acc, 0.0f);
  }
  __syncthreads();

  const int mtile = wave;
  const int rowA  = mtile * 16 + l16;  // A-matrix row for this lane

  // ---- Stage 1: x1 = relu(x0 * Wu1^T + bu1), WMMA 16x16x32 f16 ----
  for (int nt = 0; nt < 8; ++nt) {
    const int colB = nt * 16 + l16;    // B-matrix column / output column
    v8f c = {};
    #pragma unroll
    for (int kt = 0; kt < 4; ++kt) {
      const int ko = kt * 32;
      // A fragment: per-lane 8 contiguous halves at K=half*8, and at +16
      v8h alo = *(const v8h*)&sX0[rowA * H_ + ko + half * 8];
      v8h ahi = *(const v8h*)&sX0[rowA * H_ + ko + 16 + half * 8];
      // B fragment: B[K][N] = Wu1[N][K] -> 8 contiguous halves at K=half*8, and at +16
      v8h blo = *(const v8h*)&sW[colB * H_ + ko + half * 8];
      v8h bhi = *(const v8h*)&sW[colB * H_ + ko + 16 + half * 8];
      c = __builtin_amdgcn_wmma_f32_16x16x32_f16(
              false, cat8(alo, ahi), false, cat8(blo, bhi),
              (short)0, c, false, false);
    }
    const float bias = bu1[colB];
    #pragma unroll
    for (int v = 0; v < 8; ++v) {
      int m = mtile * 16 + v + 8 * half;           // C/D layout: VGPR v -> row
      sX1[m * H_ + colB] = (f16)fmaxf(c[v] + bias, 0.0f);
    }
  }
  __syncthreads();

  // ---- swap in W_e ----
  for (int i = tid; i < H_ * H_; i += 256) sW[i] = (f16)We[i];
  __syncthreads();

  // ---- Stage 2: e = x1 * We^T -> global f32 ----
  float* eB = eOut + (size_t)b * N_ * H_;
  for (int nt = 0; nt < 8; ++nt) {
    const int colB = nt * 16 + l16;
    v8f c = {};
    #pragma unroll
    for (int kt = 0; kt < 4; ++kt) {
      const int ko = kt * 32;
      v8h alo = *(const v8h*)&sX1[rowA * H_ + ko + half * 8];
      v8h ahi = *(const v8h*)&sX1[rowA * H_ + ko + 16 + half * 8];
      v8h blo = *(const v8h*)&sW[colB * H_ + ko + half * 8];
      v8h bhi = *(const v8h*)&sW[colB * H_ + ko + 16 + half * 8];
      c = __builtin_amdgcn_wmma_f32_16x16x32_f16(
              false, cat8(alo, ahi), false, cat8(blo, bhi),
              (short)0, c, false, false);
    }
    #pragma unroll
    for (int v = 0; v < 8; ++v) {
      int m = mtile * 16 + v + 8 * half;
      eB[m * H_ + colB] = c[v];
    }
  }
}

// One block per (b, j): thread = h channel; accumulate tanh over all i.
__global__ __launch_bounds__(128)
void pair_tanh_kernel(const float* __restrict__ e,    // [B,N,H]
                      const float* __restrict__ be,   // [H]
                      float* __restrict__ partial)    // [B,N,H]
{
  const int h = threadIdx.x;
  const int j = blockIdx.x;
  const int b = blockIdx.y;
  const float* eB = e + (size_t)b * N_ * H_;
  const float ej = eB[j * H_ + h] + be[h];
  float acc = 0.0f;
  #pragma unroll 4
  for (int i = 0; i < N_; ++i)
    acc += fast_tanh(ej + eB[i * H_ + h]);
  partial[((size_t)b * N_ + j) * H_ + h] = acc;
}

// Deterministic final reduce over j, scale by 1/N^2.
__global__ __launch_bounds__(128)
void reduce_kernel(const float* __restrict__ partial, // [B,N,H]
                   float* __restrict__ out)           // [B,H]
{
  const int h = threadIdx.x;
  const int b = blockIdx.x;
  float acc = 0.0f;
  #pragma unroll 4
  for (int j = 0; j < N_; ++j)
    acc += partial[((size_t)b * N_ + j) * H_ + h];
  out[b * H_ + h] = acc * (1.0f / (float)(N_ * N_));
}

extern "C" void kernel_launch(void* const* d_in, const int* in_sizes, int n_in,
                              void* d_out, int out_size, void* d_ws, size_t ws_size,
                              hipStream_t stream) {
  const float* jets = (const float*)d_in[0];
  const float* Wu   = (const float*)d_in[1];
  const float* bu   = (const float*)d_in[2];
  const float* Wu1  = (const float*)d_in[3];
  const float* bu1  = (const float*)d_in[4];
  const float* We   = (const float*)d_in[5];
  const float* be   = (const float*)d_in[6];
  float* out = (float*)d_out;

  float* e       = (float*)d_ws;            // B*N*H f32 = 4 MB
  float* partial = e + (size_t)B_ * N_ * H_; // another 4 MB

  node_mlp_kernel<<<B_, 256, 0, stream>>>(jets, Wu, bu, Wu1, bu1, We, e);
  dim3 g2(N_, B_);
  pair_tanh_kernel<<<g2, 128, 0, stream>>>(e, be, partial);
  reduce_kernel<<<B_, 128, 0, stream>>>(partial, out);
}
